// GCN_ZINC_5282809774463
// MI455X (gfx1250) — matrix operand order repaired and checked by reading, not compiled, not run on hardware
//
#include <hip/hip_runtime.h>
#include <hip/hip_bf16.h>

#define NNODES 50000
#define NEDGES 500000
#define NGRAPHS 2000
#define HID 145
#define HIDP 160            // padded to 10x16
#define NLAYERS 4
#define HMID 72             // HID // 2
#define KCHUNKS (HIDP/32)   // 5
#define NTILE (HIDP/16)     // 10
#define MTILE (NNODES/16)   // 3125 exact
#define BPK_PER_LAYER (NTILE*KCHUNKS*32*16)  // 25600 bf16 elems
#define EPSV 1e-5f

typedef __attribute__((ext_vector_type(16))) __bf16        v16bf;
typedef __attribute__((ext_vector_type(8)))  float         v8f;
typedef __attribute__((ext_vector_type(4)))  unsigned int  v4u;
typedef __attribute__((ext_vector_type(8)))  unsigned int  v8u;

__device__ __forceinline__ unsigned short f2bf_bits(float f) {
  unsigned u = __builtin_bit_cast(unsigned, f);
  unsigned r = (u + 0x7FFFu + ((u >> 16) & 1u)) >> 16;   // round-to-nearest-even
  return (unsigned short)r;
}

// ---------------- setup kernels ----------------

__global__ void init_k(float* __restrict__ dis, float* __restrict__ pooled,
                       float* __restrict__ cnt) {
  int i = blockIdx.x * blockDim.x + threadIdx.x;
  if (i < NNODES) dis[i] = 1.0f;                 // self-loop contributes 1 to degree
  if (i < NGRAPHS * HIDP) pooled[i] = 0.0f;
  if (i < NGRAPHS) cnt[i] = 0.0f;
}

__global__ void deg_k(const int* __restrict__ col, float* __restrict__ dis) {
  int e = blockIdx.x * blockDim.x + threadIdx.x;
  if (e < NEDGES) atomicAdd(&dis[col[e]], 1.0f);
}

__global__ void rsqrt_k(float* __restrict__ dis) {
  int i = blockIdx.x * blockDim.x + threadIdx.x;
  if (i < NNODES) dis[i] = rsqrtf(dis[i]);       // deg >= 1 always
}

__global__ void ew_k(const int* __restrict__ row, const int* __restrict__ col,
                     const float* __restrict__ dis, float* __restrict__ ew) {
  int e = blockIdx.x * blockDim.x + threadIdx.x;
  if (e < NEDGES) ew[e] = dis[row[e]] * dis[col[e]];
}

// Pack Wc[l] (145x145 f32) into per-lane WMMA-B-fragment bf16 layout (contiguous k order):
// Bpk[((l*NTILE+nt)*KCHUNKS+c)*32+lane][i]  = B[k = c*32 + 16*(lane/16) + i][n = nt*16 + lane%16]
__global__ void pack_b(const float* __restrict__ Wc, unsigned short* __restrict__ Bpk) {
  int idx = blockIdx.x * blockDim.x + threadIdx.x;
  if (idx >= NLAYERS * BPK_PER_LAYER) return;
  int e = idx & 15; int t = idx >> 4;
  int lane = t & 31; t >>= 5;
  int c = t % KCHUNKS; t /= KCHUNKS;
  int nt = t % NTILE;  int l = t / NTILE;
  int k = c * 32 + 16 * (lane >> 4) + e;
  int n = nt * 16 + (lane & 15);
  float v = (k < HID && n < HID) ? Wc[((size_t)l * HID + k) * HID + n] : 0.0f;
  Bpk[idx] = f2bf_bits(v);
}

__global__ void embed_k(const int* __restrict__ x, const float* __restrict__ emb,
                        float* __restrict__ h, unsigned short* __restrict__ hbf) {
  int idx = blockIdx.x * blockDim.x + threadIdx.x;
  if (idx >= NNODES * HIDP) return;
  int f = idx % HIDP, r = idx / HIDP;
  float v = (f < HID) ? emb[x[r] * HID + f] : 0.0f;
  h[idx] = v;
  hbf[idx] = f2bf_bits(v);
}

// ---------------- WMMA GEMM: hw = h_bf16 @ Wc_bf16 (f32 accumulate) ----------------
// One wave per 16-row M block; each wave computes ALL 10 N-tiles so every A
// fragment is loaded exactly once (B fragments stay hot in L0/L2: 50KB/layer).

__global__ void gemm_wmma(const unsigned short* __restrict__ Abf,  // [NNODES][HIDP] bf16
                          const unsigned short* __restrict__ Bpk,  // one layer's packed B
                          float* __restrict__ C) {                 // [NNODES][HIDP] f32
  int w = (blockIdx.x * blockDim.x + threadIdx.x) >> 5;  // wave id == M tile
  int lane = threadIdx.x & 31;
  if (w >= MTILE) return;                 // whole wave exits together (EXEC all-ones inside)
  int half = lane >> 4, mn = lane & 15;

  const unsigned short* arow = Abf + (size_t)(w * 16 + mn) * HIDP;
  v8f acc[NTILE];
#pragma unroll
  for (int tn = 0; tn < NTILE; ++tn) acc[tn] = (v8f){};

#pragma unroll
  for (int c = 0; c < KCHUNKS; ++c) {
    int kc = c * 32;
    // A fragment: elems 0..15 -> K = kc+16*half..+15
    v4u alo = *(const v4u*)(arow + kc + 16 * half);
    v4u ahi = *(const v4u*)(arow + kc + 16 * half + 8);
    v8u au;
    au[0] = alo[0]; au[1] = alo[1]; au[2] = alo[2]; au[3] = alo[3];
    au[4] = ahi[0]; au[5] = ahi[1]; au[6] = ahi[2]; au[7] = ahi[3];
    v16bf a = __builtin_bit_cast(v16bf, au);
#pragma unroll
    for (int tn = 0; tn < NTILE; ++tn) {
      // B fragment: contiguous 32B per lane (packed layout)
      const unsigned short* bp = Bpk + ((size_t)(tn * KCHUNKS + c) * 32 + lane) * 16;
      v8u bu = *(const v8u*)bp;
      v16bf b = __builtin_bit_cast(v16bf, bu);
      acc[tn] = __builtin_amdgcn_wmma_f32_16x16x32_bf16(false, a, false, b,
                                                        (short)0, acc[tn], false, false);
    }
  }
  // C layout: lane holds col n=lane%16; VGPR r holds row m = r + 8*half
#pragma unroll
  for (int tn = 0; tn < NTILE; ++tn) {
    float* crow = C + (size_t)(w * 16 + 8 * half) * HIDP + tn * 16 + mn;
#pragma unroll
    for (int r = 0; r < 8; ++r) crow[(size_t)r * HIDP] = acc[tn][r];
  }
}

// ---------------- message passing / BN / update ----------------

__global__ void agg_init_k(const float* __restrict__ bc_l, float* __restrict__ agg) {
  int idx = blockIdx.x * blockDim.x + threadIdx.x;
  if (idx >= NNODES * HIDP) return;
  int f = idx % HIDP;
  agg[idx] = (f < HID) ? bc_l[f] : 0.0f;
}

__global__ void scatter_k(const int* __restrict__ row, const int* __restrict__ col,
                          const float* __restrict__ ew, const float* __restrict__ dis,
                          const float* __restrict__ hw, float* __restrict__ agg) {
  int gw = (blockIdx.x * blockDim.x + threadIdx.x) >> 5;
  int lane = threadIdx.x & 31;
  int nw = (gridDim.x * blockDim.x) >> 5;
  for (int e = gw; e < NEDGES + NNODES; e += nw) {
    int r, c; float w;
    if (e < NEDGES) { r = row[e]; c = col[e]; w = ew[e]; }
    else { r = c = e - NEDGES; float d = dis[r]; w = d * d; }   // self loop
    const float* src = hw + (size_t)r * HIDP;
    float* dst = agg + (size_t)c * HIDP;
    for (int f = lane; f < HID; f += 32)
      atomicAdd(&dst[f], src[f] * w);
  }
}

__global__ void bn_stats_k(const float* __restrict__ agg,
                           float* __restrict__ sumb, float* __restrict__ sqb) {
  __shared__ float s1[256], s2[256];
  int f = blockIdx.x;                      // 0..HID-1
  float a = 0.f, b = 0.f;
  for (int r = threadIdx.x; r < NNODES; r += blockDim.x) {
    float v = agg[(size_t)r * HIDP + f];
    a += v; b += v * v;
  }
  s1[threadIdx.x] = a; s2[threadIdx.x] = b;
  __syncthreads();
  for (int s = 128; s > 0; s >>= 1) {
    if ((int)threadIdx.x < s) {
      s1[threadIdx.x] += s1[threadIdx.x + s];
      s2[threadIdx.x] += s2[threadIdx.x + s];
    }
    __syncthreads();
  }
  if (threadIdx.x == 0) { sumb[f] = s1[0]; sqb[f] = s2[0]; }
}

__global__ void update_k(const float* __restrict__ agg, const float* __restrict__ sumb,
                         const float* __restrict__ sqb, const float* __restrict__ gamma_l,
                         const float* __restrict__ beta_l, float* __restrict__ h,
                         unsigned short* __restrict__ hbf) {
  int idx = blockIdx.x * blockDim.x + threadIdx.x;
  if (idx >= NNODES * HIDP) return;
  int f = idx % HIDP;
  if (f < HID) {
    const float invn = 1.0f / (float)NNODES;
    float mu = sumb[f] * invn;
    float var = sqb[f] * invn - mu * mu;
    float a = agg[idx];
    float hbn = gamma_l[f] * (a - mu) * rsqrtf(var + EPSV) + beta_l[f];
    float hn = fmaxf(hbn, 0.0f) + h[idx];   // relu + residual
    h[idx] = hn;
    hbf[idx] = f2bf_bits(hn);
  } else {
    h[idx] = 0.0f;
    hbf[idx] = 0;
  }
}

// ---------------- pooling + MLP head ----------------

__global__ void pool_k(const int* __restrict__ batch, const float* __restrict__ h,
                       float* __restrict__ pooled, float* __restrict__ cnt) {
  int gw = (blockIdx.x * blockDim.x + threadIdx.x) >> 5;
  int lane = threadIdx.x & 31;
  int nw = (gridDim.x * blockDim.x) >> 5;
  for (int r = gw; r < NNODES; r += nw) {
    int g = batch[r];
    const float* src = h + (size_t)r * HIDP;
    float* dst = pooled + (size_t)g * HIDP;
    for (int f = lane; f < HID; f += 32) atomicAdd(&dst[f], src[f]);
    if (lane == 0) atomicAdd(&cnt[g], 1.0f);
  }
}

__global__ void mlp_k(const float* __restrict__ pooled, const float* __restrict__ cnt,
                      const float* __restrict__ W1, const float* __restrict__ b1,
                      const float* __restrict__ W2, const float* __restrict__ b2,
                      float* __restrict__ out) {
  int g = blockIdx.x * blockDim.x + threadIdx.x;
  if (g >= NGRAPHS) return;
  float inv = 1.0f / fmaxf(cnt[g], 1.0f);
  float hg[HID];
  for (int f = 0; f < HID; ++f) hg[f] = pooled[(size_t)g * HIDP + f] * inv;
  float o = b2[0];
  for (int j = 0; j < HMID; ++j) {
    float s = b1[j];
    for (int f = 0; f < HID; ++f) s += hg[f] * W1[f * HMID + j];
    o += fmaxf(s, 0.0f) * W2[j];
  }
  out[g] = o;
}

// ---------------- host launcher ----------------

extern "C" void kernel_launch(void* const* d_in, const int* in_sizes, int n_in,
                              void* d_out, int out_size, void* d_ws, size_t ws_size,
                              hipStream_t stream) {
  const int*   x     = (const int*)d_in[0];
  const int*   ei    = (const int*)d_in[1];
  const int*   batch = (const int*)d_in[2];
  const float* emb   = (const float*)d_in[3];
  const float* Wc    = (const float*)d_in[4];
  const float* bc    = (const float*)d_in[5];
  const float* gamma = (const float*)d_in[6];
  const float* beta  = (const float*)d_in[7];
  const float* W1    = (const float*)d_in[8];
  const float* b1    = (const float*)d_in[9];
  const float* W2    = (const float*)d_in[10];
  const float* b2    = (const float*)d_in[11];
  float* out = (float*)d_out;

  char* ws = (char*)d_ws;
  size_t off = 0;
  auto take = [&](size_t bytes) -> char* {
    char* p = ws + off;
    off = (off + bytes + 255) & ~(size_t)255;
    return p;
  };
  float*          dis    = (float*)take((size_t)NNODES * 4);
  float*          ew     = (float*)take((size_t)NEDGES * 4);
  float*          h      = (float*)take((size_t)NNODES * HIDP * 4);
  unsigned short* hbf    = (unsigned short*)take((size_t)NNODES * HIDP * 2);
  float*          hw     = (float*)take((size_t)NNODES * HIDP * 4);
  float*          agg    = (float*)take((size_t)NNODES * HIDP * 4);
  unsigned short* Bpk    = (unsigned short*)take((size_t)NLAYERS * BPK_PER_LAYER * 2);
  float*          sumb   = (float*)take(HIDP * 4);
  float*          sqb    = (float*)take(HIDP * 4);
  float*          pooled = (float*)take((size_t)NGRAPHS * HIDP * 4);
  float*          cnt    = (float*)take((size_t)NGRAPHS * 4);
  (void)ws_size; (void)in_sizes; (void)n_in; (void)out_size;

  const int* row = ei;              // edge_index[0]
  const int* col = ei + NEDGES;     // edge_index[1]

  init_k<<<(NGRAPHS * HIDP + 255) / 256, 256, 0, stream>>>(dis, pooled, cnt);
  deg_k<<<(NEDGES + 255) / 256, 256, 0, stream>>>(col, dis);
  rsqrt_k<<<(NNODES + 255) / 256, 256, 0, stream>>>(dis);
  ew_k<<<(NEDGES + 255) / 256, 256, 0, stream>>>(row, col, dis, ew);
  pack_b<<<(NLAYERS * BPK_PER_LAYER + 255) / 256, 256, 0, stream>>>(Wc, Bpk);
  embed_k<<<(NNODES * HIDP + 255) / 256, 256, 0, stream>>>(x, emb, h, hbf);

  for (int l = 0; l < NLAYERS; ++l) {
    gemm_wmma<<<(MTILE * 32 + 255) / 256, 256, 0, stream>>>(
        hbf, Bpk + (size_t)l * BPK_PER_LAYER, hw);
    agg_init_k<<<(NNODES * HIDP + 255) / 256, 256, 0, stream>>>(bc + l * HID, agg);
    scatter_k<<<2048, 256, 0, stream>>>(row, col, ew, dis, hw, agg);
    bn_stats_k<<<HID, 256, 0, stream>>>(agg, sumb, sqb);
    update_k<<<(NNODES * HIDP + 255) / 256, 256, 0, stream>>>(
        agg, sumb, sqb, gamma + l * HID, beta + l * HID, h, hbf);
  }

  pool_k<<<1024, 256, 0, stream>>>(batch, h, pooled, cnt);
  mlp_k<<<(NGRAPHS + 255) / 256, 256, 0, stream>>>(pooled, cnt, W1, b1, W2, b2, out);
}
